// LSTM_SOFT_ATT_NOX_57990648430684
// MI455X (gfx1250) — matrix operand order repaired and checked by reading, not compiled, not run on hardware
//
#include <hip/hip_runtime.h>
#include <hip/hip_bf16.h>

typedef __attribute__((ext_vector_type(16))) __bf16 v16bf;
typedef __attribute__((ext_vector_type(8)))  float  v8f;

#define B_DIM 256
#define S_DIM 196
#define R_DIM 1024
#define NT1   13          // ceil(196/16) N tiles for the attention GEMM
#define BSTR  48          // LDS row stride (bf16 elems): 96B, 32B-aligned rows

__device__ __forceinline__ unsigned short f2bf_bits(float f) {
  // native truncation: lets the backend pick v_cvt_*bf16* hardware ops
  return __builtin_bit_cast(unsigned short, (__bf16)f);
}

__device__ __forceinline__ float sigmoidf_(float x) {
  return 1.0f / (1.0f + __expf(-x));
}

// ---------------------------------------------------------------------------
// Kernel 0: att_h[b,a] = prev_h[b,:] @ Wh2a[:,a] + bh2a[a]   (256 x 196)
// ---------------------------------------------------------------------------
__global__ __launch_bounds__(256) void k_atth(
    const float* __restrict__ prev_h, const float* __restrict__ Wh2a,
    const float* __restrict__ bh2a, float* __restrict__ att_h) {
  int t = blockIdx.x * 256 + threadIdx.x;          // exactly 50176 threads
  int b = t / S_DIM;
  int a = t - b * S_DIM;
  const float* hr = prev_h + (size_t)b * R_DIM;
  float s = 0.0f;
  #pragma unroll 4
  for (int r = 0; r < R_DIM; ++r)
    s += hr[r] * Wh2a[(size_t)r * S_DIM + a];
  att_h[t] = s + bh2a[a];
}

// ---------------------------------------------------------------------------
// Kernel 1: fused  score[b,s] = sum_a tanh(att_h[b,s] + (att@Wa2a)[b,s,a]
//                                          + ba2a[a]) * Wd2d[a] + bd2d
// WMMA bf16 GEMM: M = B*S = 50176, N = 196 (pad 208), K = 1024
// Block = 128 thr (4 waves), each wave owns a 16-row M tile, all 13 N tiles.
// ---------------------------------------------------------------------------
__global__ __launch_bounds__(128) void k_score(
    const float* __restrict__ att, const float* __restrict__ Wa2a,
    const float* __restrict__ ba2a, const float* __restrict__ att_h,
    const float* __restrict__ Wd2d, const float* __restrict__ bd2d,
    float* __restrict__ score) {
  __shared__ __attribute__((aligned(32))) unsigned short ldsB[208 * BSTR];
  __shared__ float ldsS[64];

  const int tid  = threadIdx.x;
  const int lane = tid & 31;
  const int wave = tid >> 5;
  const int half = lane >> 4;        // 0: lanes 0-15, 1: lanes 16-31
  const int l16  = lane & 15;
  const int rowBase = blockIdx.x * 64 + wave * 16;   // 784*64 = 50176 exact

  if (tid < 64) ldsS[tid] = 0.0f;

  v8f acc[NT1];
  #pragma unroll
  for (int t = 0; t < NT1; ++t)
    #pragma unroll
    for (int e = 0; e < 8; ++e) acc[t][e] = 0.0f;

  for (int k0 = 0; k0 < R_DIM; k0 += 32) {
    __syncthreads();
    // Stage Wa2a chunk (32 x 208, zero-padded) into LDS as bf16, [n][k] order
    for (int idx = tid; idx < 32 * 208; idx += 128) {
      int kk = idx / 208;
      int a  = idx - kk * 208;
      float v = (a < S_DIM) ? Wa2a[(size_t)(k0 + kk) * S_DIM + a] : 0.0f;
      ldsB[a * BSTR + kk] = f2bf_bits(v);
    }
    __syncthreads();

    // Build A fragment (16x32 bf16) straight from global att rows.
    // Contiguous layout: lane<16 -> K 0..15 ; lane>=16 -> K 16..31.
    const int kb = half * 16;
    const float* arow = att + (size_t)(rowBase + l16) * R_DIM + k0;
    v16bf afrag;
    #pragma unroll
    for (int j = 0; j < 8; ++j) {
      int ks = j * 2 + kb;
      float2 f = *(const float2*)(arow + ks);
      afrag[2 * j]     = (__bf16)f.x;
      afrag[2 * j + 1] = (__bf16)f.y;
    }

    // 13 N tiles, software-pipelined B fragment loads from LDS.
    v16bf bcur = *(const v16bf*)&ldsB[(0 * 16 + l16) * BSTR + half * 16];
    #pragma unroll
    for (int t = 0; t < NT1; ++t) {
      v16bf bnext;
      if (t + 1 < NT1)
        bnext = *(const v16bf*)&ldsB[((t + 1) * 16 + l16) * BSTR + half * 16];
      acc[t] = __builtin_amdgcn_wmma_f32_16x16x32_bf16(
          false, afrag, false, bcur, (short)0, acc[t], false, false);
      if (t + 1 < NT1) bcur = bnext;
    }
  }

  // Epilogue: + ba2a + att_h, tanh, dot with Wd2d, reduce over N lanes.
  float ahr[8];
  #pragma unroll
  for (int i = 0; i < 8; ++i) ahr[i] = att_h[rowBase + half * 8 + i];

  float part[8];
  #pragma unroll
  for (int i = 0; i < 8; ++i) part[i] = 0.0f;

  #pragma unroll
  for (int t = 0; t < NT1; ++t) {
    int a = t * 16 + l16;
    if (a < S_DIM) {
      float bb = ba2a[a];
      float wd = Wd2d[a];
      #pragma unroll
      for (int i = 0; i < 8; ++i)
        part[i] += tanhf(acc[t][i] + bb + ahr[i]) * wd;
    }
  }
  #pragma unroll
  for (int i = 0; i < 8; ++i)
    atomicAdd(&ldsS[wave * 16 + half * 8 + i], part[i]);
  __syncthreads();
  if (tid < 64)
    score[blockIdx.x * 64 + tid] = ldsS[tid] + bd2d[0];
}

// ---------------------------------------------------------------------------
// Kernel 2: softmax over S per batch row
// ---------------------------------------------------------------------------
__global__ __launch_bounds__(256) void k_softmax(
    const float* __restrict__ score, float* __restrict__ weight) {
  __shared__ float red[256];
  const int b = blockIdx.x, tid = threadIdx.x;
  float x = (tid < S_DIM) ? score[b * S_DIM + tid] : -3.4e38f;
  red[tid] = x;
  __syncthreads();
  for (int o = 128; o > 0; o >>= 1) {
    if (tid < o) red[tid] = fmaxf(red[tid], red[tid + o]);
    __syncthreads();
  }
  float mx = red[0];
  __syncthreads();
  float e = (tid < S_DIM) ? __expf(x - mx) : 0.0f;
  red[tid] = e;
  __syncthreads();
  for (int o = 128; o > 0; o >>= 1) {
    if (tid < o) red[tid] += red[tid + o];
    __syncthreads();
  }
  float inv = 1.0f / red[0];
  if (tid < S_DIM) weight[b * S_DIM + tid] = e * inv;
}

// ---------------------------------------------------------------------------
// Kernel 3: att_res[b,r] = sum_s att[b,s,r] * weight[b,s]   (memory-bound)
// Block per b; 256 threads x float4 covers all 1024 r.
// ---------------------------------------------------------------------------
__global__ __launch_bounds__(256) void k_attres(
    const float* __restrict__ att, const float* __restrict__ weight,
    float* __restrict__ att_res) {
  __shared__ float lw[S_DIM];
  const int b = blockIdx.x, tid = threadIdx.x;
  if (tid < S_DIM) lw[tid] = weight[b * S_DIM + tid];
  __syncthreads();
  float4 acc = {0.0f, 0.0f, 0.0f, 0.0f};
  const float* base = att + (size_t)b * S_DIM * R_DIM + tid * 4;
  for (int s = 0; s < S_DIM; ++s) {
    float w = lw[s];
    float4 v = *(const float4*)(base + (size_t)s * R_DIM);
    acc.x += v.x * w; acc.y += v.y * w; acc.z += v.z * w; acc.w += v.w * w;
  }
  *(float4*)(att_res + (size_t)b * R_DIM + tid * 4) = acc;
}

// ---------------------------------------------------------------------------
// Kernel 4: sums = prev_h@Wh2h + att_res@Wr2a + biases, fused LSTM gates.
// Each wave: one 16x16 (m,n) tile replicated across the 4 gate columns.
// Block = 4 waves sharing one N tile (B staged once in LDS per K step).
// ---------------------------------------------------------------------------
__global__ __launch_bounds__(128) void k_lstm(
    const float* __restrict__ prev_h, const float* __restrict__ att_res,
    const float* __restrict__ Wh2h, const float* __restrict__ Wr2a,
    const float* __restrict__ bh2h, const float* __restrict__ br2a,
    const float* __restrict__ prev_c, float* __restrict__ out_c,
    float* __restrict__ out_h) {
  __shared__ __attribute__((aligned(32))) unsigned short ldsB[64 * BSTR];
  const int tid  = threadIdx.x;
  const int lane = tid & 31;
  const int wave = tid >> 5;
  const int half = lane >> 4;
  const int l16  = lane & 15;
  const int nt = blockIdx.x & 63;                 // 64 N tiles per gate
  const int mt = (blockIdx.x >> 6) * 4 + wave;    // 16 M tiles

  v8f acc[4];
  #pragma unroll
  for (int g = 0; g < 4; ++g)
    #pragma unroll
    for (int e = 0; e < 8; ++e) acc[g][e] = 0.0f;

  for (int ph = 0; ph < 2; ++ph) {
    const float* A = ph ? att_res : prev_h;
    const float* W = ph ? Wr2a : Wh2h;
    for (int k0 = 0; k0 < R_DIM; k0 += 32) {
      __syncthreads();
      // Stage 32k x (4 gates x 16n) of W into LDS as bf16, [col][k] order
      for (int idx = tid; idx < 2048; idx += 128) {
        int kk = idx >> 6;
        int cl = idx & 63;
        int g = cl >> 4, n = cl & 15;
        int col = g * R_DIM + nt * 16 + n;
        ldsB[cl * BSTR + kk] =
            f2bf_bits(W[(size_t)(k0 + kk) * (4 * R_DIM) + col]);
      }
      __syncthreads();

      const int kb = half * 16;
      const float* arow = A + (size_t)(mt * 16 + l16) * R_DIM + k0;
      v16bf afrag;
      #pragma unroll
      for (int j = 0; j < 8; ++j) {
        int ks = j * 2 + kb;
        float2 f = *(const float2*)(arow + ks);
        afrag[2 * j]     = (__bf16)f.x;
        afrag[2 * j + 1] = (__bf16)f.y;
      }
      v16bf bcur = *(const v16bf*)&ldsB[(0 * 16 + l16) * BSTR + half * 16];
      #pragma unroll
      for (int g = 0; g < 4; ++g) {
        v16bf bnext;
        if (g + 1 < 4)
          bnext = *(const v16bf*)&ldsB[((g + 1) * 16 + l16) * BSTR + half * 16];
        acc[g] = __builtin_amdgcn_wmma_f32_16x16x32_bf16(
            false, afrag, false, bcur, (short)0, acc[g], false, false);
        if (g + 1 < 4) bcur = bnext;
      }
    }
  }

  // Gate epilogue
  const int ncol = nt * 16 + l16;
  float bi = bh2h[0 * R_DIM + ncol] + br2a[0 * R_DIM + ncol];
  float bf = bh2h[1 * R_DIM + ncol] + br2a[1 * R_DIM + ncol];
  float bo = bh2h[2 * R_DIM + ncol] + br2a[2 * R_DIM + ncol];
  float bg = bh2h[3 * R_DIM + ncol] + br2a[3 * R_DIM + ncol];
  #pragma unroll
  for (int i = 0; i < 8; ++i) {
    int m = mt * 16 + half * 8 + i;
    float ig = sigmoidf_(acc[0][i] + bi);
    float fg = sigmoidf_(acc[1][i] + bf);
    float og = sigmoidf_(acc[2][i] + bo);
    float tg = tanhf(acc[3][i] + bg);
    float c = fg * prev_c[(size_t)m * R_DIM + ncol] + ig * tg;
    float h = og * tanhf(c);
    out_c[(size_t)m * R_DIM + ncol] = c;
    out_h[(size_t)m * R_DIM + ncol] = h;
  }
}

// ---------------------------------------------------------------------------
extern "C" void kernel_launch(void* const* d_in, const int* in_sizes, int n_in,
                              void* d_out, int out_size, void* d_ws,
                              size_t ws_size, hipStream_t stream) {
  (void)in_sizes; (void)n_in; (void)out_size; (void)ws_size;
  const float* att    = (const float*)d_in[0];
  const float* prev_c = (const float*)d_in[1];
  const float* prev_h = (const float*)d_in[2];
  const float* Wa2a   = (const float*)d_in[3];
  const float* ba2a   = (const float*)d_in[4];
  const float* Wh2a   = (const float*)d_in[5];
  const float* bh2a   = (const float*)d_in[6];
  const float* Wd2d   = (const float*)d_in[7];
  const float* bd2d   = (const float*)d_in[8];
  const float* Wh2h   = (const float*)d_in[9];
  const float* bh2h   = (const float*)d_in[10];
  const float* Wr2a   = (const float*)d_in[11];
  const float* br2a   = (const float*)d_in[12];

  float* ws      = (float*)d_ws;
  float* att_h   = ws;                       // 50176
  float* score   = ws + 50176;               // 50176
  float* weight  = ws + 100352;              // 50176
  float* att_res = ws + 150528;              // 262144

  float* out_c = (float*)d_out;              // 256*1024
  float* out_h = out_c + B_DIM * R_DIM;      // 256*1024

  k_atth   <<<196, 256, 0, stream>>>(prev_h, Wh2a, bh2a, att_h);
  k_score  <<<784, 128, 0, stream>>>(att, Wa2a, ba2a, att_h, Wd2d, bd2d, score);
  k_softmax<<<256, 256, 0, stream>>>(score, weight);
  k_attres <<<256, 256, 0, stream>>>(att, weight, att_res);
  k_lstm   <<<256, 128, 0, stream>>>(prev_h, att_res, Wh2h, Wr2a, bh2h, br2a,
                                     prev_c, out_c, out_h);
}
